// MS_MSA_81990925680791
// MI455X (gfx1250) — matrix operand order repaired and checked by reading, not compiled, not run on hardware
//
#include <hip/hip_runtime.h>
#include <math.h>
#include <stdint.h>

#define HGT 256
#define WW  256
#define NPIX 65536
#define CH 512
#define CSAR 2
#define HEADS 8
#define DH 64
#define EPS_BN 1e-5f
#define EPS_L2 1e-12f

typedef __attribute__((ext_vector_type(16))) __bf16 v16bf;
typedef __attribute__((ext_vector_type(8)))  float  v8f;

union AFrag { unsigned u[8]; v16bf v; };

__device__ __forceinline__ unsigned short f2bf(float f) {
  unsigned u = __float_as_uint(f);
  u += 0x7fffu + ((u >> 16) & 1u);           // round-to-nearest-even
  return (unsigned short)(u >> 16);
}
__device__ __forceinline__ float bf2f(unsigned short h) {
  return __uint_as_float(((unsigned)h) << 16);
}

// Per ISA: generic shared pointers carry the LDS byte offset in addr[31:0].
__device__ __forceinline__ unsigned lds_off(const void* p) {
  return (unsigned)(uintptr_t)p;
}
// Async global->LDS copy of 16 bytes per lane (ASYNCcnt tracked).
__device__ __forceinline__ void async_b128(unsigned lds, const void* gaddr) {
  asm volatile("global_load_async_to_lds_b128 %0, %1, off"
               :: "v"(lds), "v"(gaddr) : "memory");
}
__device__ __forceinline__ void wait_async0() {
  asm volatile("s_wait_asynccnt 0" ::: "memory");
}

// ---------------- elementwise f32 -> bf16 ----------------
__global__ void k_cvt_bf16(const float* __restrict__ s, unsigned short* __restrict__ d, int n) {
  int i = blockIdx.x * 256 + threadIdx.x;
  if (i < n) d[i] = f2bf(s[i]);
}

// ---------------- SAR branch: conv3 (2->2, 3x3, pad1) ----------------
__global__ void k_conv3(const float* __restrict__ sar, const float* __restrict__ w,
                        const float* __restrict__ b, float* __restrict__ a) {
  int p = blockIdx.x * 256 + threadIdx.x;
  int y = p >> 8, x = p & 255;
  for (int o = 0; o < CSAR; o++) {
    float acc = b[o];
    for (int ic = 0; ic < CSAR; ic++)
      for (int ky = 0; ky < 3; ky++) {
        int yy = y + ky - 1; if ((unsigned)yy >= HGT) continue;
        for (int kx = 0; kx < 3; kx++) {
          int xx = x + kx - 1; if ((unsigned)xx >= WW) continue;
          acc += w[o * 18 + ic * 9 + ky * 3 + kx] * sar[ic * NPIX + yy * WW + xx];
        }
      }
    a[o * NPIX + p] = acc;
  }
}

// ---------------- batch-norm statistics (sum, sumsq per channel) ----------------
__global__ void k_bnstats(const float* __restrict__ a, float* __restrict__ stats) {
  __shared__ float red[4][256];
  int tid = threadIdx.x;
  int stride = gridDim.x * 256;
  float s0 = 0, q0 = 0, s1 = 0, q1 = 0;
  for (int p = blockIdx.x * 256 + tid; p < NPIX; p += stride) {
    float v0 = a[p], v1 = a[NPIX + p];
    s0 += v0; q0 += v0 * v0; s1 += v1; q1 += v1 * v1;
  }
  red[0][tid] = s0; red[1][tid] = s1; red[2][tid] = q0; red[3][tid] = q1;
  __syncthreads();
  for (int s = 128; s > 0; s >>= 1) {
    if (tid < s) for (int j = 0; j < 4; j++) red[j][tid] += red[j][tid + s];
    __syncthreads();
  }
  if (tid == 0) {
    atomicAdd(&stats[0], red[0][0]); atomicAdd(&stats[1], red[1][0]);
    atomicAdd(&stats[2], red[2][0]); atomicAdd(&stats[3], red[3][0]);
  }
}

// ---------------- BN + Sobel magnitude + residual -> sar_sobel[2][N] ----------------
__global__ void k_sarsobel(const float* __restrict__ a, const float* __restrict__ stats,
                           const float* __restrict__ g, const float* __restrict__ bb,
                           const float* __restrict__ sb, float* __restrict__ ss) {
  int p = blockIdx.x * 256 + threadIdx.x;
  int y = p >> 8, x = p & 255;
  float sc[2], sh[2];
  for (int c = 0; c < 2; c++) {
    float mean = stats[c] * (1.0f / NPIX);
    float var  = stats[2 + c] * (1.0f / NPIX) - mean * mean;
    float inv  = rsqrtf(var + EPS_BN);
    sc[c] = inv * g[c];
    sh[c] = bb[c] - mean * inv * g[c];
  }
  const float we[3] = { -1.f, -2.f, -1.f };
  float ge = 0.f, go = 0.f;
  for (int ky = 0; ky < 3; ky++) {
    int yy = y + ky - 1; if ((unsigned)yy >= HGT) continue;
    for (int kx = 0; kx < 3; kx++) {
      int xx = x + kx - 1; if ((unsigned)xx >= WW) continue;
      int q = yy * WW + xx;
      float bnsum = (a[q] * sc[0] + sh[0]) + (a[NPIX + q] * sc[1] + sh[1]);
      float wev = (ky == 0) ? we[kx] : ((ky == 2) ? -we[kx] : 0.f);
      float wod = (kx == 0) ? we[ky] : ((kx == 2) ? -we[ky] : 0.f);
      ge += wev * bnsum; go += wod * bnsum;
    }
  }
  float s0 = ge + sb[0], s1 = go + sb[1], s2 = ge + sb[2], s3 = go + sb[3];
  float sob0 = sqrtf(s0 * s0 + s1 * s1), sob1 = sqrtf(s2 * s2 + s3 * s3);
  float bn0 = a[p] * sc[0] + sh[0], bn1 = a[NPIX + p] * sc[1] + sh[1];
  ss[p] = sob0 + bn0;
  ss[NPIX + p] = sob1 + bn1;
}

// ---------------- e1 = 1x1 conv (2 -> 512), layout [pix][c] ----------------
__global__ void k_e1(const float* __restrict__ ss, const float* __restrict__ w2,
                     const float* __restrict__ b2, float* __restrict__ e1) {
  long t = (long)blockIdx.x * 256 + threadIdx.x;
  int c = (int)(t & (CH - 1));
  int p = (int)(t >> 9);
  e1[t] = w2[c * 2] * ss[p] + w2[c * 2 + 1] * ss[NPIX + p] + b2[c];
}

// ---------------- mask = e1 + sigmoid(conv3x3(ss)) + dw5x5(e1) ----------------
__global__ void k_mask(const float* __restrict__ ss, const float* __restrict__ e1,
                       const float* __restrict__ w32, const float* __restrict__ b32,
                       const float* __restrict__ wd, const float* __restrict__ bd,
                       float* __restrict__ mask) {
  long t = (long)blockIdx.x * 256 + threadIdx.x;
  int c = (int)(t & (CH - 1));
  int p = (int)(t >> 9);
  int y = p >> 8, x = p & 255;
  float e2 = b32[c];
  for (int ky = 0; ky < 3; ky++) {
    int yy = y + ky - 1; if ((unsigned)yy >= HGT) continue;
    for (int kx = 0; kx < 3; kx++) {
      int xx = x + kx - 1; if ((unsigned)xx >= WW) continue;
      int q = yy * WW + xx;
      e2 += w32[c * 18 + ky * 3 + kx] * ss[q] + w32[c * 18 + 9 + ky * 3 + kx] * ss[NPIX + q];
    }
  }
  e2 = 1.0f / (1.0f + expf(-e2));
  float e3 = bd[c];
  for (int ky = 0; ky < 5; ky++) {
    int yy = y + ky - 2; if ((unsigned)yy >= HGT) continue;
    for (int kx = 0; kx < 5; kx++) {
      int xx = x + kx - 2; if ((unsigned)xx >= WW) continue;
      e3 += wd[c * 25 + ky * 5 + kx] * e1[((long)(yy * WW + xx)) * CH + c];
    }
  }
  mask[t] = e1[t] + e2 + e3;
}

// ---------------- generic bf16 WMMA GEMM: C[M,N] = A[M,K] * B[K,N] (+bias) ----------------
// Block tile 128 x BN (BN = 128 or 64), 8 waves.
//   BN=128: wave tile 32x64 (2x4 WMMA tiles, 8 v_wmma per K-step)
//   BN= 64: wave tile 32x32 (2x2 WMMA tiles)
// A tile (128x32 bf16) staged with global_load_async_to_lds_b128 (ASYNCcnt);
// B tile staged K-transposed with VALU so bf16 K-pairs are contiguous for the B frag layout.
// LDS row stride = 40 ushorts (80B): 16B-aligned for b128 async writes, bank-conflict-free.
template<int BN>
__global__ __launch_bounds__(256) void k_gemm(
    const unsigned short* __restrict__ A, int lda,
    const unsigned short* __restrict__ B, int ldb,
    float* __restrict__ Cf, unsigned short* __restrict__ Cb,
    const float* __restrict__ bias, int ldc, int K) {
  constexpr int NT = BN / 32;                 // WMMA tiles per wave in N
  __shared__ unsigned short As[128 * 40];
  __shared__ unsigned short Bs[BN * 40];
  const int tid = threadIdx.x;
  const int m0 = blockIdx.x * 128;
  const int n0 = blockIdx.y * BN;
  const unsigned* Bu = (const unsigned*)B;
  const int ldbu = ldb >> 1;
  const int wid = tid >> 5, lane = tid & 31;
  const int lane15 = lane & 15, half = lane >> 4;
  const int wm = (wid & 3) * 32, wn = (wid >> 2) * (BN / 2);

  v8f acc[2][NT];
  for (int i = 0; i < 2; i++) for (int j = 0; j < NT; j++) for (int r = 0; r < 8; r++) acc[i][j][r] = 0.f;

  const int arow = tid >> 1;                  // A staging: each thread copies 32B of one row
  const int acol = (tid & 1) * 16;            // ushort offset within the 32-wide K chunk

  for (int k0 = 0; k0 < K; k0 += 32) {
    { // stage A asynchronously: 128 rows x 64B
      const unsigned short* g = A + (long)(m0 + arow) * lda + k0 + acol;
      unsigned l = lds_off(&As[arow * 40 + acol]);
      async_b128(l, g);
      async_b128(l + 16, (const char*)g + 16);
    }
    for (int i = 0; i < BN / 16; i++) {       // stage B transposed: [n][k]
      int idx = tid * (BN / 16) + i;
      int kr = idx >> (BN == 128 ? 6 : 5), nu = idx & (BN / 2 - 1);
      unsigned gv = Bu[(long)(k0 + kr) * ldbu + (n0 >> 1) + nu];
      Bs[(2 * nu) * 40 + kr]     = (unsigned short)(gv & 0xffffu);
      Bs[(2 * nu + 1) * 40 + kr] = (unsigned short)(gv >> 16);
    }
    if (k0 + 32 < K)                          // L2 prefetch of next A slab
      __builtin_prefetch(A + (long)(m0 + arow) * lda + (k0 + 32) + acol, 0, 1);
    wait_async0();
    __syncthreads();

    AFrag a0, a1;
    for (int v = 0; v < 8; v++) {
      int uidx = (v < 4) ? (half * 4 + v) : (8 + half * 4 + (v - 4));   // 16-bit A layout
      int ka = 2 * uidx;
      a0.u[v] = *(const unsigned*)&As[(wm + lane15) * 40 + ka];
      a1.u[v] = *(const unsigned*)&As[(wm + 16 + lane15) * 40 + ka];
    }
    for (int nt = 0; nt < NT; nt++) {
      AFrag b;
      for (int v = 0; v < 8; v++) {                                     // 16-bit B layout
        int kb = (v < 4) ? (half * 8 + 2 * v) : (16 + half * 8 + 2 * (v - 4));
        b.u[v] = *(const unsigned*)&Bs[(wn + nt * 16 + lane15) * 40 + kb];
      }
      acc[0][nt] = __builtin_amdgcn_wmma_f32_16x16x32_bf16(false, a0.v, false, b.v, (short)0, acc[0][nt], false, false);
      acc[1][nt] = __builtin_amdgcn_wmma_f32_16x16x32_bf16(false, a1.v, false, b.v, (short)0, acc[1][nt], false, false);
    }
    __syncthreads();
  }

  for (int mt = 0; mt < 2; mt++)
    for (int nt = 0; nt < NT; nt++) {
      int mbase = m0 + wm + mt * 16 + half * 8;       // 32-bit C/D layout
      int n = n0 + wn + nt * 16 + lane15;
      float bv = bias ? bias[n] : 0.f;
      for (int r = 0; r < 8; r++) {
        float val = acc[mt][nt][r] + bv;
        long offc = (long)(mbase + r) * ldc + n;
        if (Cf) Cf[offc] = val;
        if (Cb) Cb[offc] = f2bf(val);
      }
    }
}

// ---------------- per-channel L2 norms of q (blocks 0..511) and k (512..1023) ----------------
__global__ void k_norms(const unsigned short* __restrict__ q,
                        const unsigned short* __restrict__ k,
                        float* __restrict__ norms) {
  __shared__ float red[256];
  int b = blockIdx.x;
  int ch = b & 511;
  const unsigned short* src = (b >= 512) ? k : q;
  float s = 0.f;
  for (int p = threadIdx.x; p < NPIX; p += 256) {
    float v = bf2f(src[(long)p * CH + ch]);
    s += v * v;
  }
  red[threadIdx.x] = s; __syncthreads();
  for (int st = 128; st > 0; st >>= 1) {
    if (threadIdx.x < st) red[threadIdx.x] += red[threadIdx.x + st];
    __syncthreads();
  }
  if (threadIdx.x == 0) norms[b] = sqrtf(red[0]);
}

// ---------------- attn_raw[h][i][j] += sum_n k[n,hi]*q[n,hj]  (split-K WMMA + f32 atomics) ----------------
__global__ __launch_bounds__(512) void k_attn(const unsigned short* __restrict__ kbf,
                                              const unsigned short* __restrict__ qbf,
                                              float* __restrict__ attn_raw) {
  __shared__ unsigned short Kt[64 * 34];
  __shared__ unsigned short Qt[64 * 34];
  const int h = blockIdx.y;
  const int n0 = blockIdx.x * 4096;
  const int tid = threadIdx.x, wid = tid >> 5, lane = tid & 31;
  const int lane15 = lane & 15, half = lane >> 4;
  const int ti = wid >> 2, tj = wid & 3;     // 16 waves cover the 64x64 tile grid
  const unsigned* Ku = (const unsigned*)kbf;
  const unsigned* Qu = (const unsigned*)qbf;
  v8f acc; for (int r = 0; r < 8; r++) acc[r] = 0.f;

  for (int ks = 0; ks < 4096; ks += 32) {
    for (int i = 0; i < 2; i++) {            // stage [ch][tok] transposed chunks
      int idx = tid * 2 + i;
      int tok = idx >> 5, cu = idx & 31;
      long goff = (long)(n0 + ks + tok) * (CH / 2) + (h * DH / 2) + cu;
      unsigned gk = Ku[goff], gq = Qu[goff];
      Kt[(2 * cu) * 34 + tok]     = (unsigned short)(gk & 0xffffu);
      Kt[(2 * cu + 1) * 34 + tok] = (unsigned short)(gk >> 16);
      Qt[(2 * cu) * 34 + tok]     = (unsigned short)(gq & 0xffffu);
      Qt[(2 * cu + 1) * 34 + tok] = (unsigned short)(gq >> 16);
    }
    __syncthreads();
    AFrag a, bq;
    for (int v = 0; v < 8; v++) {
      int uidx = (v < 4) ? (half * 4 + v) : (8 + half * 4 + (v - 4));
      a.u[v]  = *(const unsigned*)&Kt[(ti * 16 + lane15) * 34 + 2 * uidx];
      int kb = 2 * uidx;
      bq.u[v] = *(const unsigned*)&Qt[(tj * 16 + lane15) * 34 + kb];
    }
    acc = __builtin_amdgcn_wmma_f32_16x16x32_bf16(false, a.v, false, bq.v, (short)0, acc, false, false);
    __syncthreads();
  }
  int i0 = ti * 16 + half * 8, j = tj * 16 + lane15;
  for (int r = 0; r < 8; r++)
    atomicAdd(&attn_raw[h * 4096 + (i0 + r) * 64 + j], acc[r]);
}

// ---------------- softmax over j with l2-norm scaling; write attn^T in bf16 ----------------
__global__ void k_softmax(const float* __restrict__ attn_raw, const float* __restrict__ norms,
                          const float* __restrict__ rescale, unsigned short* __restrict__ attnT) {
  __shared__ float red[64];
  int row = blockIdx.x;
  int h = row >> 6, i = row & 63, j = threadIdx.x;
  float nk = fmaxf(norms[512 + h * 64 + i], EPS_L2);
  float nq = fmaxf(norms[h * 64 + j], EPS_L2);
  float l = attn_raw[h * 4096 + i * 64 + j] * rescale[h] / (nk * nq);
  red[j] = l; __syncthreads();
  for (int s = 32; s > 0; s >>= 1) { if (j < s) red[j] = fmaxf(red[j], red[j + s]); __syncthreads(); }
  float mx = red[0]; __syncthreads();
  float e = expf(l - mx);
  red[j] = e; __syncthreads();
  for (int s = 32; s > 0; s >>= 1) { if (j < s) red[j] += red[j + s]; __syncthreads(); }
  attnT[h * 4096 + j * 64 + i] = f2bf(e / red[0]);
}

// ---------------- vm = bf16(v * mask) ----------------
__global__ void k_vm(const float* __restrict__ v, const float* __restrict__ mask,
                     unsigned short* __restrict__ vm) {
  long t = (long)blockIdx.x * 256 + threadIdx.x;
  vm[t] = f2bf(v[t] * mask[t]);
}

// ---------------- PE branch: dw3x3 then exact GELU ----------------
__global__ void k_pe1(const float* __restrict__ v, const float* __restrict__ w,
                      float* __restrict__ o) {
  long t = (long)blockIdx.x * 256 + threadIdx.x;
  int c = (int)(t & (CH - 1));
  int p = (int)(t >> 9);
  int y = p >> 8, x = p & 255;
  float s = 0.f;
  for (int ky = 0; ky < 3; ky++) {
    int yy = y + ky - 1; if ((unsigned)yy >= HGT) continue;
    for (int kx = 0; kx < 3; kx++) {
      int xx = x + kx - 1; if ((unsigned)xx >= WW) continue;
      s += w[c * 9 + ky * 3 + kx] * v[((long)(yy * WW + xx)) * CH + c];
    }
  }
  o[t] = 0.5f * s * (1.0f + erff(s * 0.70710678118654752f));
}

// ---------------- dw3x3 of PE temp, accumulated into d_out ----------------
__global__ void k_pe2_add(const float* __restrict__ pe, const float* __restrict__ w,
                          float* __restrict__ out) {
  long t = (long)blockIdx.x * 256 + threadIdx.x;
  int c = (int)(t & (CH - 1));
  int p = (int)(t >> 9);
  int y = p >> 8, x = p & 255;
  float s = 0.f;
  for (int ky = 0; ky < 3; ky++) {
    int yy = y + ky - 1; if ((unsigned)yy >= HGT) continue;
    for (int kx = 0; kx < 3; kx++) {
      int xx = x + kx - 1; if ((unsigned)xx >= WW) continue;
      s += w[c * 9 + ky * 3 + kx] * pe[((long)(yy * WW + xx)) * CH + c];
    }
  }
  out[t] += s;
}

extern "C" void kernel_launch(void* const* d_in, const int* in_sizes, int n_in,
                              void* d_out, int out_size, void* d_ws, size_t ws_size,
                              hipStream_t stream) {
  (void)in_sizes; (void)n_in; (void)out_size; (void)ws_size;
  const float* x_in   = (const float*)d_in[0];
  const float* sar    = (const float*)d_in[1];
  const float* Wq     = (const float*)d_in[2];
  const float* Wk     = (const float*)d_in[3];
  const float* Wv     = (const float*)d_in[4];
  const float* rescale= (const float*)d_in[5];
  const float* Wp     = (const float*)d_in[6];
  const float* bp     = (const float*)d_in[7];
  const float* pe1w   = (const float*)d_in[8];
  const float* pe2w   = (const float*)d_in[9];
  const float* c3w    = (const float*)d_in[10];
  const float* c3b    = (const float*)d_in[11];
  const float* bng    = (const float*)d_in[12];
  const float* bnb    = (const float*)d_in[13];
  const float* sob_b  = (const float*)d_in[14];
  const float* c2w    = (const float*)d_in[15];
  const float* c2b    = (const float*)d_in[16];
  const float* c32w   = (const float*)d_in[17];
  const float* c32b   = (const float*)d_in[18];
  const float* dcw    = (const float*)d_in[19];
  const float* dcb    = (const float*)d_in[20];
  float* out = (float*)d_out;

  const size_t NC = (size_t)NPIX * CH;
  char* base = (char*)d_ws;
  size_t off = 0;
  auto alloc = [&](size_t bytes) -> void* {
    void* p = base + off;
    off += (bytes + 255) & ~(size_t)255;
    return p;
  };
  unsigned short* x_bf   = (unsigned short*)alloc(NC * 2);
  unsigned short* q_bf   = (unsigned short*)alloc(NC * 2);
  unsigned short* k_bf   = (unsigned short*)alloc(NC * 2);
  unsigned short* vm_bf  = (unsigned short*)alloc(NC * 2);
  unsigned short* o_bf   = (unsigned short*)alloc(NC * 2);
  float* v_f             = (float*)alloc(NC * 4);
  float* e1_f            = (float*)alloc(NC * 4);   // reused later as PE temp
  float* mask_f          = (float*)alloc(NC * 4);
  unsigned short* wq_bf  = (unsigned short*)alloc((size_t)CH * CH * 2);
  unsigned short* wk_bf  = (unsigned short*)alloc((size_t)CH * CH * 2);
  unsigned short* wv_bf  = (unsigned short*)alloc((size_t)CH * CH * 2);
  unsigned short* wp_bf  = (unsigned short*)alloc((size_t)CH * CH * 2);
  float* a_conv          = (float*)alloc((size_t)2 * NPIX * 4);
  float* sarso           = (float*)alloc((size_t)2 * NPIX * 4);
  float* stats           = (float*)alloc(256);
  float* norms           = (float*)alloc(1024 * 4);
  float* attn_raw        = (float*)alloc((size_t)HEADS * 64 * 64 * 4);
  unsigned short* attnT  = (unsigned short*)alloc((size_t)HEADS * 64 * 64 * 2);

  const int NWORD = (int)NC;                 // 33.5M
  const int GRID_NC = NWORD / 256;           // 131072
  const int GRID_W  = (CH * CH) / 256;       // 1024

  // precision conversions
  k_cvt_bf16<<<GRID_NC, 256, 0, stream>>>(x_in, x_bf, NWORD);
  k_cvt_bf16<<<GRID_W, 256, 0, stream>>>(Wq, wq_bf, CH * CH);
  k_cvt_bf16<<<GRID_W, 256, 0, stream>>>(Wk, wk_bf, CH * CH);
  k_cvt_bf16<<<GRID_W, 256, 0, stream>>>(Wv, wv_bf, CH * CH);
  k_cvt_bf16<<<GRID_W, 256, 0, stream>>>(Wp, wp_bf, CH * CH);

  // SAR-guided mask
  k_conv3<<<NPIX / 256, 256, 0, stream>>>(sar, c3w, c3b, a_conv);
  hipMemsetAsync(stats, 0, 16, stream);
  k_bnstats<<<256, 256, 0, stream>>>(a_conv, stats);
  k_sarsobel<<<NPIX / 256, 256, 0, stream>>>(a_conv, stats, bng, bnb, sob_b, sarso);
  k_e1<<<GRID_NC, 256, 0, stream>>>(sarso, c2w, c2b, e1_f);
  k_mask<<<GRID_NC, 256, 0, stream>>>(sarso, e1_f, c32w, c32b, dcw, dcb, mask_f);

  // QKV projections (WMMA, 128x128 block tiles)
  dim3 gg(NPIX / 128, CH / 128);
  k_gemm<128><<<gg, 256, 0, stream>>>(x_bf, CH, wq_bf, CH, nullptr, q_bf, nullptr, CH, CH);
  k_gemm<128><<<gg, 256, 0, stream>>>(x_bf, CH, wk_bf, CH, nullptr, k_bf, nullptr, CH, CH);
  k_gemm<128><<<gg, 256, 0, stream>>>(x_bf, CH, wv_bf, CH, v_f, nullptr, nullptr, CH, CH);
  k_vm<<<GRID_NC, 256, 0, stream>>>(v_f, mask_f, vm_bf);

  // channel attention
  k_norms<<<1024, 256, 0, stream>>>(q_bf, k_bf, norms);
  hipMemsetAsync(attn_raw, 0, (size_t)HEADS * 64 * 64 * 4, stream);
  k_attn<<<dim3(16, HEADS), 512, 0, stream>>>(k_bf, q_bf, attn_raw);
  k_softmax<<<HEADS * 64, 64, 0, stream>>>(attn_raw, norms, rescale, attnT);

  // o = vm * attn^T (per-head K=64 WMMA GEMM, 128x64 block tiles)
  for (int h = 0; h < HEADS; h++) {
    k_gemm<64><<<dim3(NPIX / 128, 1), 256, 0, stream>>>(
        vm_bf + h * DH, CH, attnT + h * 4096, DH,
        nullptr, o_bf + h * DH, nullptr, CH, DH);
  }

  // out_c = o @ Wp + bp  -> d_out
  k_gemm<128><<<gg, 256, 0, stream>>>(o_bf, CH, wp_bf, CH, out, nullptr, bp, CH, CH);

  // positional branch: dw3x3 -> GELU -> dw3x3, add into d_out (e1 buffer reused as temp)
  k_pe1<<<GRID_NC, 256, 0, stream>>>(v_f, pe1w, e1_f);
  k_pe2_add<<<GRID_NC, 256, 0, stream>>>(e1_f, pe2w, out);
}
